// VectorQuantizer_71021579207266
// MI455X (gfx1250) — compile-verified, matrix operand order checked by reading, not run on hardware
//
#include <hip/hip_runtime.h>
#include <hip/hip_bf16.h>

typedef __attribute__((ext_vector_type(16))) __bf16 v16bf;
typedef __attribute__((ext_vector_type(8)))  float  v8f;

#define D_DIM   512
#define K_CODES 4096
#define N_ROWS  16384
#define MT      128   // z rows per workgroup
#define NTC     128   // codebook rows per chunk
#define KB      64    // K per LDS stage
#define PITCH   72    // LDS row pitch in bf16 elems (64 + 8 pad, keeps 16B align)

// ---------- helpers ----------

__device__ inline unsigned pk_bf16(float a, float b) {
    unsigned ua = __float_as_uint(a), ub = __float_as_uint(b);
    ua = (ua + 0x7FFFu + ((ua >> 16) & 1u)) >> 16;      // RNE truncate
    ub = (ub + 0x7FFFu + ((ub >> 16) & 1u)) >> 16;
    return ua | (ub << 16);
}

struct FragBits { uint4 lo, hi; };

// A & B fragments share the same per-lane addressing on CDNA5 (both operands
// are K-contiguous row-major: z rows for A, codebook rows for B columns):
// lane = hi*16+lo ; row = lo ; elems 0..7 = K kk*32 + hi*8 .. +7 ; elems 8..15 = +16
__device__ inline v16bf load_frag(const unsigned short* lds, int row, int kk, int hi) {
    const unsigned short* base = lds + row * PITCH + kk * 32 + hi * 8;
    FragBits fb;
    fb.lo = *(const uint4*)(base);
    fb.hi = *(const uint4*)(base + 16);
    return __builtin_bit_cast(v16bf, fb);
}

// ---------- kernel 1: zero ws accumulators ----------

__global__ __launch_bounds__(256) void vq_zero_kernel(float* counts, double* sumsq) {
    int i = blockIdx.x * 256 + threadIdx.x;
    if (i < K_CODES) counts[i] = 0.0f;
    if (i == 0) *sumsq = 0.0;
}

// ---------- kernel 2: one-pass fp32 -> bf16 conversion (precompute path) ----------

__global__ __launch_bounds__(256) void vq_cvt_bf16_kernel(const float* __restrict__ in,
                                                          unsigned* __restrict__ out, int n4) {
    int stride = gridDim.x * 256;
    for (int i = blockIdx.x * 256 + threadIdx.x; i < n4; i += stride) {
        float4 f = ((const float4*)in)[i];
        ((uint2*)out)[i] = make_uint2(pk_bf16(f.x, f.y), pk_bf16(f.z, f.w));
    }
}

// ---------- kernel 3: codebook squared norms (one wave per code row) ----------

__global__ __launch_bounds__(256) void vq_cbnorm_kernel(const float* __restrict__ cb,
                                                        float* __restrict__ enorm) {
    int row  = blockIdx.x * 8 + (threadIdx.x >> 5);
    int lane = threadIdx.x & 31;
    const float* r = cb + (size_t)row * D_DIM;
    float s = 0.0f;
    for (int j = lane; j < D_DIM; j += 32) { float x = r[j]; s += x * x; }
    #pragma unroll
    for (int off = 16; off >= 1; off >>= 1) s += __shfl_xor(s, off, 32);
    if (lane == 0) enorm[row] = s;
}

// ---------- kernel 4: WMMA distance GEMM + running argmin ----------
// PRE=true : zp/cbp are bf16 (ushort) arrays, LDS fill is pure B128 copy.
// PRE=false: zp/cbp are fp32 arrays, convert on the fly (small-ws fallback).

template <bool PRE>
__global__ __launch_bounds__(256) void vq_argmin_kernel(const void* __restrict__ zp,
                                                        const void* __restrict__ cbp,
                                                        const float* __restrict__ enorm,
                                                        int* __restrict__ out_idx) {
    __shared__ unsigned short ldsA[MT  * PITCH];   // 18 KB
    __shared__ unsigned short ldsB[NTC * PITCH];   // 18 KB

    const int tid  = threadIdx.x;
    const int wave = tid >> 5;        // 8 waves, each owns a 16-row M strip
    const int lane = tid & 31;
    const int lo   = lane & 15;
    const int hi   = lane >> 4;
    const int mblk = blockIdx.x * MT;

    float minv[8];
    int   mini[8];
    #pragma unroll
    for (int v = 0; v < 8; ++v) { minv[v] = 3.4e38f; mini[v] = 0; }

    for (int nc = 0; nc < K_CODES; nc += NTC) {
        v8f acc[8];
        #pragma unroll
        for (int nt = 0; nt < 8; ++nt)
            #pragma unroll
            for (int e = 0; e < 8; ++e) acc[nt][e] = 0.0f;

        for (int kb = 0; kb < D_DIM; kb += KB) {
            __syncthreads();   // protect previous stage's LDS reads
            if (PRE) {
                // pure copy: 128x64 bf16 tiles, 16B granules, zero conversion VALU
                const unsigned short* zb  = (const unsigned short*)zp;
                const unsigned short* cbb = (const unsigned short*)cbp;
                #pragma unroll
                for (int i = 0; i < 4; ++i) {
                    int linear = tid + i * 256;      // uint4 slot id, 0..1023
                    int r  = linear >> 3;
                    int c8 = linear & 7;
                    uint4 va = *(const uint4*)(zb  + (size_t)(mblk + r) * D_DIM + kb + c8 * 8);
                    uint4 vb = *(const uint4*)(cbb + (size_t)(nc   + r) * D_DIM + kb + c8 * 8);
                    *(uint4*)&ldsA[r * PITCH + c8 * 8] = va;
                    *(uint4*)&ldsB[r * PITCH + c8 * 8] = vb;
                }
            } else {
                // load fp32 and convert on the fly
                const float* zf  = (const float*)zp;
                const float* cbf = (const float*)cbp;
                #pragma unroll
                for (int i = 0; i < 8; ++i) {
                    int linear = tid + i * 256;      // float4 slot id, 0..2047
                    int r  = linear >> 4;
                    int c4 = linear & 15;
                    float4 fa = ((const float4*)(zf  + (size_t)(mblk + r) * D_DIM + kb))[c4];
                    float4 fb = ((const float4*)(cbf + (size_t)(nc   + r) * D_DIM + kb))[c4];
                    *(uint2*)&ldsA[r * PITCH + c4 * 4] = make_uint2(pk_bf16(fa.x, fa.y), pk_bf16(fa.z, fa.w));
                    *(uint2*)&ldsB[r * PITCH + c4 * 4] = make_uint2(pk_bf16(fb.x, fb.y), pk_bf16(fb.z, fb.w));
                }
            }
            __syncthreads();

            #pragma unroll
            for (int kk = 0; kk < 2; ++kk) {                 // 2 x K=32 steps
                v16bf aF = load_frag(ldsA, wave * 16 + lo, kk, hi);
                #pragma unroll
                for (int nt = 0; nt < 8; ++nt) {
                    v16bf bF = load_frag(ldsB, nt * 16 + lo, kk, hi);
                    acc[nt] = __builtin_amdgcn_wmma_f32_16x16x32_bf16(
                        false, aF, false, bF, (short)0, acc[nt], false, false);
                }
            }
        }

        // distance score = ||e||^2 - 2*dot  (||z||^2 constant per row -> irrelevant to argmin)
        #pragma unroll
        for (int nt = 0; nt < 8; ++nt) {
            int   code = nc + nt * 16 + lo;
            float en   = enorm[code];
            #pragma unroll
            for (int v = 0; v < 8; ++v) {
                float s = en - 2.0f * acc[nt][v];
                if (s < minv[v]) { minv[v] = s; mini[v] = code; }  // ascending scan -> first-min
            }
        }
    }

    // reduce argmin across the 16 lanes holding different code columns
    #pragma unroll
    for (int v = 0; v < 8; ++v) {
        float bv = minv[v];
        int   bi = mini[v];
        #pragma unroll
        for (int off = 8; off >= 1; off >>= 1) {
            float ov = __shfl_xor(bv, off, 32);
            int   oi = __shfl_xor(bi, off, 32);
            if (ov < bv || (ov == bv && oi < bi)) { bv = ov; bi = oi; }
        }
        // lane 0 -> local row v ; lane 16 -> local row v+8
        if (lo == 0) out_idx[mblk + wave * 16 + hi * 8 + v] = bi;
    }
}

// ---------- kernel 5: gather z_q, STE output, loss partial, histogram ----------

__global__ __launch_bounds__(256) void vq_gather_kernel(const float* __restrict__ z,
                                                        const float* __restrict__ cb,
                                                        const int* __restrict__ idx,
                                                        float* __restrict__ out_zq,
                                                        float* __restrict__ out_idx_f,
                                                        float* __restrict__ counts,
                                                        double* __restrict__ sumsq) {
    int row  = blockIdx.x * 8 + (threadIdx.x >> 5);
    int lane = threadIdx.x & 31;
    int k    = idx[row];
    const float4* zr = (const float4*)(z  + (size_t)row * D_DIM);
    const float4* cr = (const float4*)(cb + (size_t)k   * D_DIM);
    float4*       o  = (float4*)(out_zq + (size_t)row * D_DIM);
    float s = 0.0f;
    #pragma unroll
    for (int j = 0; j < 4; ++j) {
        int p = lane + j * 32;
        float4 zv = zr[p], cv = cr[p], d, ov;
        d.x = cv.x - zv.x; d.y = cv.y - zv.y; d.z = cv.z - zv.z; d.w = cv.w - zv.w;
        s += d.x * d.x + d.y * d.y + d.z * d.z + d.w * d.w;
        ov.x = zv.x + d.x; ov.y = zv.y + d.y; ov.z = zv.z + d.z; ov.w = zv.w + d.w; // STE numerics
        o[p] = ov;
    }
    #pragma unroll
    for (int off = 16; off >= 1; off >>= 1) s += __shfl_xor(s, off, 32);
    if (lane == 0) {
        atomicAdd(sumsq, (double)s);
        atomicAdd(&counts[k], 1.0f);
        out_idx_f[row] = (float)k;
    }
}

// ---------- kernel 6: finalize loss + perplexity ----------

__global__ __launch_bounds__(256) void vq_final_kernel(const float* __restrict__ counts,
                                                       const double* __restrict__ sumsq,
                                                       float* __restrict__ out_loss,
                                                       float* __restrict__ out_perp) {
    __shared__ float red[256];
    int t = threadIdx.x;
    float s = 0.0f;
    for (int k = t; k < K_CODES; k += 256) {
        float p = counts[k] * (1.0f / (float)N_ROWS);
        s += p * logf(p + 1e-10f);
    }
    red[t] = s;
    __syncthreads();
    for (int st = 128; st >= 1; st >>= 1) {
        if (t < st) red[t] += red[t + st];
        __syncthreads();
    }
    if (t == 0) {
        *out_perp = expf(-red[0]);
        *out_loss = 0.25f * (float)(*sumsq / (double)((double)N_ROWS * (double)D_DIM));
    }
}

// ---------- launch ----------

extern "C" void kernel_launch(void* const* d_in, const int* in_sizes, int n_in,
                              void* d_out, int out_size, void* d_ws, size_t ws_size,
                              hipStream_t stream) {
    const float* z  = (const float*)d_in[0];   // [8,2048,512] fp32
    const float* cb = (const float*)d_in[1];   // [4096,512]   fp32

    char*   ws     = (char*)d_ws;
    int*    ws_idx = (int*)ws;                         // 16384 * 4 B @ 0
    float*  counts = (float*)(ws + 65536);             //  4096 * 4 B
    float*  enorm  = (float*)(ws + 65536 + 16384);     //  4096 * 4 B
    double* sumsq  = (double*)(ws + 98304);            //  8 B

    const size_t z_elems  = (size_t)N_ROWS * D_DIM;    // 8388608
    const size_t cb_elems = (size_t)K_CODES * D_DIM;   // 2097152
    unsigned short* z_bf  = (unsigned short*)(ws + 131072);
    unsigned short* cb_bf = (unsigned short*)(ws + 131072 + z_elems * 2);
    const size_t ws_need  = 131072 + (z_elems + cb_elems) * 2;   // ~20.1 MB

    float* out       = (float*)d_out;
    float* out_zq    = out;                              // 8388608 elems
    float* out_loss  = out + z_elems;                    // 1
    float* out_idx_f = out_loss + 1;                     // 16384 (indices as float)
    float* out_perp  = out_idx_f + N_ROWS;               // 1

    vq_zero_kernel<<<16, 256, 0, stream>>>(counts, sumsq);
    vq_cbnorm_kernel<<<K_CODES / 8, 256, 0, stream>>>(cb, enorm);

    if (ws_size >= ws_need) {
        // precompute bf16 copies once; main loop's LDS fill is conversion-free
        vq_cvt_bf16_kernel<<<2048, 256, 0, stream>>>(z,  (unsigned*)z_bf,  (int)(z_elems / 4));
        vq_cvt_bf16_kernel<<<1024, 256, 0, stream>>>(cb, (unsigned*)cb_bf, (int)(cb_elems / 4));
        vq_argmin_kernel<true><<<N_ROWS / MT, 256, 0, stream>>>(
            (const void*)z_bf, (const void*)cb_bf, enorm, ws_idx);
    } else {
        vq_argmin_kernel<false><<<N_ROWS / MT, 256, 0, stream>>>(
            (const void*)z, (const void*)cb, enorm, ws_idx);
    }

    vq_gather_kernel<<<N_ROWS / 8, 256, 0, stream>>>(z, cb, ws_idx, out_zq, out_idx_f, counts, sumsq);
    vq_final_kernel<<<1, 256, 0, stream>>>(counts, sumsq, out_loss, out_perp);
}
